// MT_63376537419953
// MI455X (gfx1250) — matrix operand order repaired and checked by reading, not compiled, not run on hardware
//
#include <hip/hip_runtime.h>
#include <hip/hip_bf16.h>
#include <math.h>

#define DD   120
#define KPAD 128

typedef __attribute__((ext_vector_type(16))) _Float16 v16h;
typedef __attribute__((ext_vector_type(8)))  _Float16 v8h;
typedef __attribute__((ext_vector_type(4)))  _Float16 v4h;
typedef __attribute__((ext_vector_type(8)))  float    v8f;
typedef __attribute__((ext_vector_type(4)))  float    v4f;

__device__ __forceinline__ float sigf(float x) { return 1.f / (1.f + expf(-x)); }

// ---------------------------------------------------------------------------
// Weight prep: W [Nout,Kin] f32 row-major -> f16 zero-padded [Npad][128]
// ---------------------------------------------------------------------------
__global__ void wprep_kernel(const float* __restrict__ W, _Float16* __restrict__ out,
                             int Nout, int Kin, int total) {
    int i = blockIdx.x * 256 + threadIdx.x;
    if (i >= total) return;
    int n = i >> 7, k = i & 127;
    float v = (n < Nout && k < Kin) ? W[n * Kin + k] : 0.f;
    out[i] = (_Float16)v;
}

// Bias prep: zero-padded to tile grid
__global__ void bprep_kernel(const float* __restrict__ b, float* __restrict__ out,
                             int Nout, int total) {
    int i = blockIdx.x * 256 + threadIdx.x;
    if (i >= total) return;
    out[i] = (i < Nout) ? b[i] : 0.f;
}

// ---------------------------------------------------------------------------
// Fused WMMA GEMM:  out[m, 0:NT*16] = act(A)[m,:] @ W.T + bias   (K padded 128)
// AMODE: 0 = plain A, 1 = tanh(A), 2 = gather rows: A_row(i) = gsrc[gidx[i]]
// LDA / LDOUT compile-time so all loads/stores use immediate offsets.
// Block = 256 threads = 8 waves; 128 rows per block.
// ---------------------------------------------------------------------------
template <int NT, int AMODE, int LDOUT, int LDA>
__global__ __launch_bounds__(256)
void gemm_wmma_kernel(const float* __restrict__ A,
                      const _Float16* __restrict__ W,
                      const float* __restrict__ bias_pad,
                      float* __restrict__ out,
                      int nrows,
                      const int* __restrict__ gidx, const float* __restrict__ gsrc)
{
    __shared__ _Float16 sA[128 * KPAD];   // 32 KB

    const int r0  = blockIdx.x * 128;
    const int tid = threadIdx.x;

    // Stage + convert A tile: b128 global loads -> b64 DS stores.
    for (int e = tid * 4; e < 128 * KPAD; e += 256 * 4) {
        int r = e >> 7, k = e & 127;
        int row = r0 + r;
        v4f v = {0.f, 0.f, 0.f, 0.f};
        if (row < nrows && k < DD) {
            const float* rp;
            if (AMODE == 2) rp = gsrc + (size_t)gidx[row] * LDA;
            else            rp = A + (size_t)row * LDA;
            v = *(const v4f*)(rp + k);
            if (AMODE == 1) {
                v.x = tanhf(v.x); v.y = tanhf(v.y); v.z = tanhf(v.z); v.w = tanhf(v.w);
            }
        }
        v4h hv; hv[0] = (_Float16)v.x; hv[1] = (_Float16)v.y;
        hv[2] = (_Float16)v.z; hv[3] = (_Float16)v.w;
        *(v4h*)&sA[e] = hv;
    }
    __syncthreads();

    const int wave = tid >> 5;
    const int lane = tid & 31;
    const int h    = lane >> 4;
    const int ln   = lane & 15;

    // A fragments (ISA 16-bit A layout): row M = ln,
    //   j=0..7 -> K = c*32 + h*8 + j ; j=8..15 -> K = c*32 + 16 + h*8 + (j-8)
    const int rowbase = (wave * 16 + ln) * KPAD;
    v16h afrag[4];
    #pragma unroll
    for (int c = 0; c < 4; ++c) {
        union { v16h v; v8h p[2]; } u;
        u.p[0] = *(const v8h*)&sA[rowbase + c * 32 + h * 8];
        u.p[1] = *(const v8h*)&sA[rowbase + c * 32 + 16 + h * 8];
        afrag[c] = u.v;
    }

    const bool full = (r0 + 128) <= nrows;

    // Shared bases; all per-tile/per-chunk displacements become immediates.
    const _Float16* wb = W + (size_t)ln * KPAD + h * 8;           // + t*16*KPAD + c*32
    const float*    bb = bias_pad + ln;                           // + t*16
    float*          ob = out + (size_t)(r0 + wave * 16 + h * 8) * LDOUT + ln;  // + t*16 + r*LDOUT

    #pragma unroll
    for (int t = 0; t < NT; ++t) {
        const float bv = bb[t * 16];
        v8f acc;
        #pragma unroll
        for (int r = 0; r < 8; ++r) acc[r] = bv;

        v16h bf[4];
        #pragma unroll
        for (int c = 0; c < 4; ++c) {
            union { v16h v; v8h p[2]; } ub;
            ub.p[0] = *(const v8h*)(wb + t * 16 * KPAD + c * 32);
            ub.p[1] = *(const v8h*)(wb + t * 16 * KPAD + c * 32 + 16);
            bf[c] = ub.v;
        }
        #pragma unroll
        for (int c = 0; c < 4; ++c)
            acc = __builtin_amdgcn_wmma_f32_16x16x32_f16(
                false, afrag[c], false, bf[c], (short)0, acc, false, false);

        // C/D layout: VGPR r -> M = r + 8*h, col = t*16 + ln
        if (full) {
            #pragma unroll
            for (int r = 0; r < 8; ++r) ob[t * 16 + r * LDOUT] = acc[r];
        } else {
            #pragma unroll
            for (int r = 0; r < 8; ++r) {
                int m = r0 + wave * 16 + h * 8 + r;
                if (m < nrows) ob[t * 16 + r * LDOUT] = acc[r];
            }
        }
    }
}

// ---------------------------------------------------------------------------
// Segment helpers (seg arrays sorted -> per-graph row ranges, no atomics)
// ---------------------------------------------------------------------------
__device__ __forceinline__ int lower_bound_dev(const int* a, int n, int v) {
    int lo = 0, hi = n;
    while (lo < hi) { int m = (lo + hi) >> 1; if (a[m] < v) lo = m + 1; else hi = m; }
    return lo;
}

__global__ __launch_bounds__(128)
void seg_avg_kernel(const float* __restrict__ s, const int* __restrict__ seg,
                    int N, float* __restrict__ avg) {
    int g = blockIdx.x;
    __shared__ int sb, se;
    if (threadIdx.x == 0) { sb = lower_bound_dev(seg, N, g); se = lower_bound_dev(seg, N, g + 1); }
    __syncthreads();
    int d = threadIdx.x;
    if (d >= DD) return;
    float sum = 0.f;
    for (int i = sb; i < se; ++i) sum += s[(size_t)i * KPAD + d];
    avg[g * DD + d] = sum / fmaxf((float)(se - sb), 1.f);
}

// att = sigmoid(relu(avg @ w1.T) @ w2.T)   w1:[15,120] w2:[120,15]
__global__ __launch_bounds__(128)
void se_kernel(const float* __restrict__ avg, const float* __restrict__ w1,
               const float* __restrict__ w2, float* __restrict__ att) {
    int g = blockIdx.x, tid = threadIdx.x;
    __shared__ float ar[DD];
    __shared__ float tt[15];
    if (tid < DD) ar[tid] = avg[g * DD + tid];
    __syncthreads();
    if (tid < 15) {
        float a = 0.f;
        for (int k = 0; k < DD; ++k) a += ar[k] * w1[tid * DD + k];
        tt[tid] = fmaxf(a, 0.f);
    }
    __syncthreads();
    if (tid < DD) {
        float a = 0.f;
        for (int j = 0; j < 15; ++j) a += tt[j] * w2[tid * 15 + j];
        att[g * DD + tid] = sigf(a);
    }
}

// rowsum[i] = sum_d s[i,d] * att[seg[i],d]   (one wave per row)
__global__ __launch_bounds__(256)
void rowsum_kernel(const float* __restrict__ s, const float* __restrict__ att,
                   const int* __restrict__ seg, int N, float* __restrict__ rowsum) {
    int wave = threadIdx.x >> 5, lane = threadIdx.x & 31;
    int row = blockIdx.x * 8 + wave;
    if (row >= N) return;
    int g = seg[row];
    float p = 0.f;
    for (int k = lane; k < DD; k += 32) p += s[(size_t)row * KPAD + k] * att[g * DD + k];
    for (int off = 16; off; off >>= 1) p += __shfl_down(p, off, 32);
    if (lane == 0) rowsum[row] = p;
}

// gpool[g,d] = (sum_i s[i,d]*rowsum[i] / 120) * sum_i sigmoid(s[i,d])
__global__ __launch_bounds__(128)
void gpool_kernel(const float* __restrict__ s, const float* __restrict__ rowsum,
                  const int* __restrict__ seg, int N, float* __restrict__ gpool) {
    int g = blockIdx.x;
    __shared__ int sb, se;
    if (threadIdx.x == 0) { sb = lower_bound_dev(seg, N, g); se = lower_bound_dev(seg, N, g + 1); }
    __syncthreads();
    int d = threadIdx.x;
    if (d >= DD) return;
    float s1 = 0.f, s2 = 0.f;
    for (int i = sb; i < se; ++i) {
        float v = s[(size_t)i * KPAD + d];
        s1 += v * rowsum[i];
        s2 += sigf(v);
    }
    gpool[g * DD + d] = (s1 / (float)DD) * s2;
}

// h = z*b + (1-z)*a2, z = sigmoid(a2+b); in place over a2 buffer (padded cols = 0)
__global__ void hmix_kernel(float* __restrict__ hbuf, const float* __restrict__ b, long n) {
    long i = (long)blockIdx.x * 256 + threadIdx.x;
    if (i >= n) return;
    float a2 = hbuf[i], bb = b[i];
    float z = sigf(a2 + bb);
    hbuf[i] = z * bb + (1.f - z) * a2;
}

// GRU finish: gx/gh stride 384, h stride 128, out stride 120
__global__ void gru_out_kernel(const float* __restrict__ gx, const float* __restrict__ gh,
                               const float* __restrict__ hbuf, float* __restrict__ out, int N) {
    long i = (long)blockIdx.x * 256 + threadIdx.x;
    if (i >= (long)N * DD) return;
    int m = (int)(i / DD), d = (int)(i % DD);
    size_t b3 = (size_t)m * 384;
    float r  = sigf(gx[b3 + d] + gh[b3 + d]);
    float zz = sigf(gx[b3 + DD + d] + gh[b3 + DD + d]);
    float nn = tanhf(gx[b3 + 2 * DD + d] + r * gh[b3 + 2 * DD + d]);
    out[i] = (1.f - zz) * nn + zz * hbuf[(size_t)m * KPAD + d];
}

// ---------------------------------------------------------------------------
extern "C" void kernel_launch(void* const* d_in, const int* in_sizes, int n_in,
                              void* d_out, int out_size, void* d_ws, size_t ws_size,
                              hipStream_t stream) {
    const float* va     = (const float*)d_in[0];
    const float* vb     = (const float*)d_in[1];
    const float* B_w    = (const float*)d_in[2];
    const float* B_b    = (const float*)d_in[3];
    const float* se_w1  = (const float*)d_in[4];
    const float* se_w2  = (const float*)d_in[5];
    const float* E_w    = (const float*)d_in[6];
    const float* E_b    = (const float*)d_in[7];
    const float* GA_w   = (const float*)d_in[8];
    const float* GA_b   = (const float*)d_in[9];
    const float* GB_w   = (const float*)d_in[10];
    const float* GB_b   = (const float*)d_in[11];
    const float* wih    = (const float*)d_in[12];
    const float* whh    = (const float*)d_in[13];
    const float* bih    = (const float*)d_in[14];
    const float* bhh    = (const float*)d_in[15];
    const int*   seg_a  = (const int*)d_in[16];
    const int*   seg_b  = (const int*)d_in[17];

    const int Na = in_sizes[0] / DD;
    const int Nb = in_sizes[1] / DD;
    const int G  = 2048;   // fixed by reference setup

    char* wsb = (char*)d_ws;
    size_t off = 0;
    auto alloc = [&](size_t bytes) -> void* {
        void* p = wsb + off;
        off = (off + bytes + 255) & ~(size_t)255;
        return p;
    };
    float* s_ws   = (float*)alloc((size_t)Na * KPAD * 4);   // ld 128
    float* vblin  = (float*)alloc((size_t)Nb * KPAD * 4);   // ld 128
    float* b_ws   = (float*)alloc((size_t)Nb * KPAD * 4);   // ld 128
    float* h_ws   = (float*)alloc((size_t)Nb * KPAD * 4);   // a2 then h, ld 128
    float* gx_ws  = (float*)alloc((size_t)Nb * 384 * 4);    // ld 384
    float* gh_ws  = (float*)alloc((size_t)Nb * 384 * 4);    // ld 384
    float* rowsum = (float*)alloc((size_t)Na * 4);
    float* avg    = (float*)alloc((size_t)G * DD * 4);
    float* att    = (float*)alloc((size_t)G * DD * 4);
    float* gpool  = (float*)alloc((size_t)G * DD * 4);
    _Float16* Bw16  = (_Float16*)alloc(128 * KPAD * 2);
    _Float16* Ew16  = (_Float16*)alloc(128 * KPAD * 2);
    _Float16* GAw16 = (_Float16*)alloc(128 * KPAD * 2);
    _Float16* GBw16 = (_Float16*)alloc(128 * KPAD * 2);
    _Float16* wih16 = (_Float16*)alloc(384 * KPAD * 2);
    _Float16* whh16 = (_Float16*)alloc(384 * KPAD * 2);
    float* Bb_p   = (float*)alloc(128 * 4);
    float* Eb_p   = (float*)alloc(128 * 4);
    float* GAb_p  = (float*)alloc(128 * 4);
    float* GBb_p  = (float*)alloc(128 * 4);
    float* bih_p  = (float*)alloc(384 * 4);
    float* bhh_p  = (float*)alloc(384 * 4);

    const int t128 = 128 * KPAD, t384 = 384 * KPAD;
    wprep_kernel<<<(t128 + 255) / 256, 256, 0, stream>>>(B_w,  Bw16,  DD,  DD, t128);
    wprep_kernel<<<(t128 + 255) / 256, 256, 0, stream>>>(E_w,  Ew16,  DD,  DD, t128);
    wprep_kernel<<<(t128 + 255) / 256, 256, 0, stream>>>(GA_w, GAw16, DD,  DD, t128);
    wprep_kernel<<<(t128 + 255) / 256, 256, 0, stream>>>(GB_w, GBw16, DD,  DD, t128);
    wprep_kernel<<<(t384 + 255) / 256, 256, 0, stream>>>(wih,  wih16, 360, DD, t384);
    wprep_kernel<<<(t384 + 255) / 256, 256, 0, stream>>>(whh,  whh16, 360, DD, t384);
    bprep_kernel<<<1, 256, 0, stream>>>(B_b,  Bb_p,  DD, 128);
    bprep_kernel<<<1, 256, 0, stream>>>(E_b,  Eb_p,  DD, 128);
    bprep_kernel<<<1, 256, 0, stream>>>(GA_b, GAb_p, DD, 128);
    bprep_kernel<<<1, 256, 0, stream>>>(GB_b, GBb_p, DD, 128);
    bprep_kernel<<<2, 256, 0, stream>>>(bih,  bih_p, 360, 384);
    bprep_kernel<<<2, 256, 0, stream>>>(bhh,  bhh_p, 360, 384);

    const int blkA = (Na + 127) / 128;
    const int blkB = (Nb + 127) / 128;

    // s = va @ B_w.T + B_b
    gemm_wmma_kernel<8, 0, KPAD, DD><<<blkA, 256, 0, stream>>>(va, Bw16, Bb_p, s_ws, Na, nullptr, nullptr);

    // segment mean -> SE attention -> per-row sums -> gpool
    seg_avg_kernel<<<G, 128, 0, stream>>>(s_ws, seg_a, Na, avg);
    se_kernel<<<G, 128, 0, stream>>>(avg, se_w1, se_w2, att);
    rowsum_kernel<<<(Na + 7) / 8, 256, 0, stream>>>(s_ws, att, seg_a, Na, rowsum);
    gpool_kernel<<<G, 128, 0, stream>>>(s_ws, rowsum, seg_a, Na, gpool);

    // vb_lin = vb @ E.T + E_b
    gemm_wmma_kernel<8, 0, KPAD, DD><<<blkB, 256, 0, stream>>>(vb, Ew16, Eb_p, vblin, Nb, nullptr, nullptr);
    // b = tanh(vb_lin) @ GB.T + GB_b
    gemm_wmma_kernel<8, 1, KPAD, KPAD><<<blkB, 256, 0, stream>>>(vblin, GBw16, GBb_p, b_ws, Nb, nullptr, nullptr);
    // a2 = gpool[seg_b] @ GA.T + GA_b
    gemm_wmma_kernel<8, 2, KPAD, DD><<<blkB, 256, 0, stream>>>(gpool, GAw16, GAb_p, h_ws, Nb, seg_b, gpool);
    // h = z*b + (1-z)*a2
    hmix_kernel<<<(int)(((long)Nb * KPAD + 255) / 256), 256, 0, stream>>>(h_ws, b_ws, (long)Nb * KPAD);
    // gx = vb_lin @ wih.T + bih ; gh = h @ whh.T + bhh
    gemm_wmma_kernel<23, 0, 384, KPAD><<<blkB, 256, 0, stream>>>(vblin, wih16, bih_p, gx_ws, Nb, nullptr, nullptr);
    gemm_wmma_kernel<23, 0, 384, KPAD><<<blkB, 256, 0, stream>>>(h_ws, whh16, bhh_p, gh_ws, Nb, nullptr, nullptr);
    // GRU combine -> output
    gru_out_kernel<<<(int)(((long)Nb * DD + 255) / 256), 256, 0, stream>>>(gx_ws, gh_ws, h_ws, (float*)d_out, Nb);
}
